// CrossAttention_83391085019674
// MI455X (gfx1250) — compile-verified, matrix operand order checked by reading, not run on hardware
//
#include <hip/hip_runtime.h>
#include <hip/hip_bf16.h>
#include <stdint.h>

// ---------------------------------------------------------------------------
// Types for CDNA5 WMMA (wave32): bf16 16x16x32, f32 accumulate.
// ---------------------------------------------------------------------------
typedef __bf16   bf16x16 __attribute__((ext_vector_type(16)));
typedef __bf16   bf16x4  __attribute__((ext_vector_type(4)));
typedef float    f32x8   __attribute__((ext_vector_type(8)));
typedef unsigned u32x4   __attribute__((ext_vector_type(4)));
typedef int      i32x4   __attribute__((ext_vector_type(4)));
typedef int      i32x8   __attribute__((ext_vector_type(8)));

__device__ __forceinline__ f32x8 wmma_bf16(bf16x16 a, bf16x16 b, f32x8 c) {
  return __builtin_amdgcn_wmma_f32_16x16x32_bf16(
      /*neg_a=*/false, a, /*neg_b=*/false, b,
      /*c_mod=*/(short)0, c, /*reuse_a=*/false, /*reuse_b=*/false);
}

// Load one 16x32 bf16 fragment row for this lane. Memory holds 32 contiguous
// K-values per row; per the CDNA5 16-bit A/B layout, lanes 0-15 take K 0..7
// and 16..23, lanes 16-31 take K 8..15 and 24..31 (two 16-byte chunks).
__device__ __forceinline__ bf16x16 load_frag(const __bf16* p, int half) {
  union { bf16x16 v; u32x4 q[2]; } u;
  u.q[0] = *(const u32x4*)(p + half * 8);
  u.q[1] = *(const u32x4*)(p + 16 + half * 8);
  return u.v;
}

// ---------------------------------------------------------------------------
// Tensor Data Mover: async DMA of a [rows x 32] bf16 tile (row-major, row
// stride = rowStride elems) from global into LDS, packed contiguous.
// Descriptor packing per CDNA5 ISA.
// ---------------------------------------------------------------------------
#if defined(__gfx1250__)
__device__ __forceinline__ void tdm_load_tile_bf16(unsigned lds_off,
                                                   const __bf16* gptr,
                                                   unsigned tileRows,
                                                   unsigned tileCols,
                                                   unsigned rowStride) {
  unsigned long long ga = (unsigned long long)(uintptr_t)gptr;
  u32x4 g0;
  g0.x = 1u;                                   // count=1 (valid), user mode
  g0.y = lds_off;                              // lds_addr (bytes)
  g0.z = (unsigned)ga;                         // global_addr[31:0]
  g0.w = (unsigned)((ga >> 32) & 0x1FFFFFFull) // global_addr[56:32]
         | (2u << 30);                         // type=2 ("image")
  i32x8 g1;
  g1[0] = (int)(1u << 16);                     // wg_mask=0, data_size=1 (2B)
  g1[1] = (int)((rowStride & 0xFFFFu) << 16);  // tensor_dim0[15:0]
  g1[2] = (int)(((rowStride >> 16) & 0xFFFFu)  // tensor_dim0[31:16]
         | (0xFFFFu << 16));                   // tensor_dim1[15:0] (large)
  g1[3] = (int)(0x0000FFFFu                    // tensor_dim1[31:16] (large)
         | (tileCols << 16));                  // tile_dim0
  g1[4] = (int)(tileRows & 0xFFFFu);           // tile_dim1 (tile_dim2=0)
  g1[5] = (int)rowStride;                      // tensor_dim0_stride[31:0]
  g1[6] = 0;                                   // stride hi / dim1_stride lo
  g1[7] = 0;
  i32x4 gz = {0, 0, 0, 0};
#if __clang_major__ >= 23
  i32x8 gz8 = {0, 0, 0, 0, 0, 0, 0, 0};
  __builtin_amdgcn_tensor_load_to_lds(g0, g1, gz, gz, gz8, 0);
#else
  __builtin_amdgcn_tensor_load_to_lds(g0, g1, gz, gz, 0);
#endif
}
#endif

// ---------------------------------------------------------------------------
// Elementwise f32 -> bf16 convert (b128 load, single b64 packed store).
// ---------------------------------------------------------------------------
__global__ __launch_bounds__(256) void cvt_f32_to_bf16(
    const float* __restrict__ src, __bf16* __restrict__ dst, long long n4) {
  long long i = (long long)blockIdx.x * 256 + threadIdx.x;
  const long long stride = (long long)gridDim.x * 256;
  for (; i < n4; i += stride) {
    float4 v = ((const float4*)src)[i];
    bf16x4 o;
    o.x = (__bf16)v.x; o.y = (__bf16)v.y; o.z = (__bf16)v.z; o.w = (__bf16)v.w;
    ((bf16x4*)dst)[i] = o;
  }
}

// ---------------------------------------------------------------------------
// features [B][P][D] f32 -> featT [B][D][P] bf16 (LDS-tiled 32x32 transpose)
// ---------------------------------------------------------------------------
__global__ __launch_bounds__(256) void transpose_cvt_bf16(
    const float* __restrict__ feat, __bf16* __restrict__ featT, int P, int D) {
  __shared__ float tile[32][33];
  const int b = blockIdx.z;
  const int p0 = blockIdx.y * 32;
  const int d0 = blockIdx.x * 32;
  const int tx = threadIdx.x & 31;
  const int ty = threadIdx.x >> 5;  // 8 row-groups
  const float* src = feat + ((long long)b * P + p0) * D + d0;
  for (int r = ty; r < 32; r += 8) tile[r][tx] = src[(long long)r * D + tx];
  __syncthreads();
  __bf16* dst = featT + ((long long)b * D + d0) * P + p0;
  for (int r = ty; r < 32; r += 8) dst[(long long)r * P + tx] = (__bf16)tile[tx][r];
}

// ---------------------------------------------------------------------------
// NT GEMM: C_bf16[M][N] = alpha * (A_bf16[M][K] @ W_bf16[N][K]^T) + bias[N]
// Block tile 128x128, K-step 32, 8 waves (4x2 over M x N), TDM double-buffered
// LDS staging, 8 WMMAs per wave per K-step. M, N, K all multiples of tile.
// ---------------------------------------------------------------------------
#define GBM 128
#define GBN 128
#define GBK 32

__global__ __launch_bounds__(256) void gemm_nt_bias_bf16(
    const __bf16* __restrict__ A, const __bf16* __restrict__ W,
    const float* __restrict__ bias, __bf16* __restrict__ C,
    int N, int K, float alpha) {
  __shared__ __attribute__((aligned(16))) __bf16 Asm[2][GBM * GBK];
  __shared__ __attribute__((aligned(16))) __bf16 Bsm[2][GBN * GBK];

  const int tid = threadIdx.x;
  const int lane = tid & 31;
  const int wv = tid >> 5;    // 0..7
  const int wm = wv >> 1;     // 0..3 -> 32 rows each
  const int wn = wv & 1;      // 0..1 -> 64 cols each
  const int half = lane >> 4; // 0/1
  const int l16 = lane & 15;

  const long long rowA0 = (long long)blockIdx.y * GBM;
  const long long rowB0 = (long long)blockIdx.x * GBN;

  f32x8 acc[2][4];
#pragma unroll
  for (int i = 0; i < 2; ++i)
#pragma unroll
    for (int j = 0; j < 4; ++j) acc[i][j] = (f32x8)(0.0f);

  const int ksteps = K / GBK;

#if defined(__gfx1250__)
  if (wv == 0) {  // TDM issues once per wave (EXEC ignored)
    tdm_load_tile_bf16((unsigned)(uintptr_t)&Asm[0][0], A + rowA0 * K, GBM, GBK, (unsigned)K);
    tdm_load_tile_bf16((unsigned)(uintptr_t)&Bsm[0][0], W + rowB0 * K, GBN, GBK, (unsigned)K);
    __builtin_amdgcn_s_wait_tensorcnt(0);
  }
#endif
  __syncthreads();

  for (int ks = 0; ks < ksteps; ++ks) {
    const int cur = ks & 1;
    const int nxt = cur ^ 1;
#if defined(__gfx1250__)
    if (wv == 0 && (ks + 1) < ksteps) {  // prefetch next tiles into other buffer
      tdm_load_tile_bf16((unsigned)(uintptr_t)&Asm[nxt][0],
                         A + rowA0 * K + (long long)(ks + 1) * GBK, GBM, GBK, (unsigned)K);
      tdm_load_tile_bf16((unsigned)(uintptr_t)&Bsm[nxt][0],
                         W + rowB0 * K + (long long)(ks + 1) * GBK, GBN, GBK, (unsigned)K);
    }
#endif
    bf16x16 af[2], bf[4];
#pragma unroll
    for (int i = 0; i < 2; ++i)
      af[i] = load_frag(&Asm[cur][(wm * 32 + i * 16 + l16) * GBK], half);
#pragma unroll
    for (int j = 0; j < 4; ++j)
      bf[j] = load_frag(&Bsm[cur][(wn * 64 + j * 16 + l16) * GBK], half);
#pragma unroll
    for (int i = 0; i < 2; ++i)
#pragma unroll
      for (int j = 0; j < 4; ++j) acc[i][j] = wmma_bf16(af[i], bf[j], acc[i][j]);

#if defined(__gfx1250__)
    if (wv == 0) __builtin_amdgcn_s_wait_tensorcnt(0);
#endif
    __syncthreads();
  }

  // Epilogue: C layout — VGPR r holds M = r + 8*(lane>=16), N = lane&15.
  // Grid exactly tiles M and N: no bounds checks, 32-bit local indexing.
#pragma unroll
  for (int i = 0; i < 2; ++i) {
    const long long gm0 = rowA0 + (wm * 32 + i * 16 + half * 8);
#pragma unroll
    for (int j = 0; j < 4; ++j) {
      const int gn = (int)rowB0 + wn * 64 + j * 16 + l16;
      const float bv = bias[gn];
      __bf16* cp = C + gm0 * N + gn;
#pragma unroll
      for (int r = 0; r < 8; ++r) cp[r * N] = (__bf16)(acc[i][j][r] * alpha + bv);
    }
  }
}

// ---------------------------------------------------------------------------
// Attention: per block = one batch x 16 query rows.
//  P1: S[16][576] = Qtile @ K^T (WMMA over K=4096; waves 0-3 own 5 N-tiles,
//      waves 4-7 own 4, fully unrolled)
//  P2: row softmax in LDS (wave-level shuffle reductions), cast to bf16
//  P3: out[16][4096] = attn @ featT^T (NT via pre-transposed features)
// ---------------------------------------------------------------------------
#define TQ 16
#define PDIM 576
#define NTP (PDIM / 16)  // 36
#define KSP (PDIM / 32)  // 18

__global__ __launch_bounds__(256) void cross_attention_core(
    const __bf16* __restrict__ Q,   // [B*T][D]  (pre-scaled by 1/sqrt(D))
    const __bf16* __restrict__ Km,  // [B*P][D]
    const __bf16* __restrict__ Vt,  // [B][D][P] (features transposed)
    float* __restrict__ out,        // [B*T][D]
    int T, int D) {
  __shared__ __attribute__((aligned(16))) float  S[TQ][PDIM];
  __shared__ __attribute__((aligned(16))) __bf16 Abf[TQ][PDIM];

  const int b = blockIdx.y;
  const int t0 = blockIdx.x * TQ;
  const int tid = threadIdx.x;
  const int lane = tid & 31;
  const int wv = tid >> 5;
  const int half = lane >> 4;
  const int l16 = lane & 15;
  const bool has5 = (wv < 4);  // wave-uniform: tiles 32..35 exist only here

  const __bf16* Qb = Q + ((long long)b * T + t0) * D;
  const __bf16* Kb = Km + (long long)b * PDIM * D;

  // ---- Phase 1: scores ----
  f32x8 acc[4], acc4;
#pragma unroll
  for (int j = 0; j < 4; ++j) acc[j] = (f32x8)(0.0f);
  acc4 = (f32x8)(0.0f);

  const __bf16* qrow = Qb + (long long)l16 * D;
  const __bf16* krow0 = Kb + (long long)(wv * 16 + l16) * D;  // tile wv
  const long long ktile = 128LL * D;                          // 8 tiles apart
  for (int ks = 0; ks < D / 32; ++ks) {
    __builtin_prefetch(qrow + (ks + 1) * 32, 0, 0);  // global_prefetch_b8
    const bf16x16 af = load_frag(qrow + ks * 32, half);
#pragma unroll
    for (int j = 0; j < 4; ++j)
      acc[j] = wmma_bf16(af, load_frag(krow0 + j * ktile + ks * 32, half), acc[j]);
    if (has5)
      acc4 = wmma_bf16(af, load_frag(krow0 + 4 * ktile + ks * 32, half), acc4);
  }
#pragma unroll
  for (int j = 0; j < 4; ++j) {
    const int nt = wv + 8 * j;
#pragma unroll
    for (int r = 0; r < 8; ++r) S[r + 8 * half][nt * 16 + l16] = acc[j][r];
  }
  if (has5) {
    const int nt = wv + 32;
#pragma unroll
    for (int r = 0; r < 8; ++r) S[r + 8 * half][nt * 16 + l16] = acc4[r];
  }
  __syncthreads();

  // ---- Phase 2: softmax (wave w owns rows 2w, 2w+1; 576 = 18*32 cols) ----
  for (int rr = 0; rr < 2; ++rr) {
    const int row = wv * 2 + rr;
    float m = -3.4e38f;
    for (int c = lane; c < PDIM; c += 32) m = fmaxf(m, S[row][c]);
#pragma unroll
    for (int off = 16; off; off >>= 1) m = fmaxf(m, __shfl_xor(m, off, 32));
    float s = 0.0f;
    for (int c = lane; c < PDIM; c += 32) {
      const float e = __expf(S[row][c] - m);
      S[row][c] = e;
      s += e;
    }
#pragma unroll
    for (int off = 16; off; off >>= 1) s += __shfl_xor(s, off, 32);
    const float inv = 1.0f / s;
    for (int c = lane; c < PDIM; c += 32) Abf[row][c] = (__bf16)(S[row][c] * inv);
  }
  __syncthreads();

  // ---- Phase 3: attended = attn @ features (wave w owns 512 output cols) ----
  const __bf16* Vb = Vt + (long long)b * D * PDIM;
  float* Ob = out + ((long long)b * T + t0) * D;
  const int colbase = wv * 512;
  for (int ch = 0; ch < 16; ++ch) {
    const int c0 = colbase + ch * 32;
    f32x8 a2[2];
    a2[0] = (f32x8)(0.0f);
    a2[1] = (f32x8)(0.0f);
    for (int ks = 0; ks < KSP; ++ks) {
      const bf16x16 af = load_frag(&Abf[l16][ks * 32], half);
#pragma unroll
      for (int fn = 0; fn < 2; ++fn) {
        const __bf16* vrow = Vb + (long long)(c0 + fn * 16 + l16) * PDIM + ks * 32;
        a2[fn] = wmma_bf16(af, load_frag(vrow, half), a2[fn]);
      }
    }
#pragma unroll
    for (int fn = 0; fn < 2; ++fn)
#pragma unroll
      for (int r = 0; r < 8; ++r)
        Ob[(long long)(r + 8 * half) * D + c0 + fn * 16 + l16] = a2[fn][r];
  }
}

// ---------------------------------------------------------------------------
// Host launcher
// ---------------------------------------------------------------------------
extern "C" void kernel_launch(void* const* d_in, const int* in_sizes, int n_in,
                              void* d_out, int out_size, void* d_ws, size_t ws_size,
                              hipStream_t stream) {
  (void)in_sizes; (void)n_in; (void)out_size; (void)ws_size;
  const float* text = (const float*)d_in[0];
  const float* feats = (const float*)d_in[1];
  const float* Wq = (const float*)d_in[2];
  const float* bq = (const float*)d_in[3];
  const float* Wk = (const float*)d_in[4];
  const float* bk = (const float*)d_in[5];
  float* out = (float*)d_out;

  const int B = 8, T = 2048, P = 576, D = 4096;
  const long long nText = (long long)B * T * D;  // 67,108,864
  const long long nFeat = (long long)B * P * D;  // 18,874,368
  const long long nW = (long long)D * D;         // 16,777,216

  char* ws = (char*)d_ws;
  size_t off = 0;
  auto take = [&](long long elems) -> __bf16* {
    __bf16* p = (__bf16*)(ws + off);
    off += ((size_t)elems * 2 + 255) & ~(size_t)255;
    return p;
  };
  __bf16* text16  = take(nText);
  __bf16* feat16  = take(nFeat);
  __bf16* Wq16    = take(nW);
  __bf16* Wk16    = take(nW);
  __bf16* featT16 = take(nFeat);
  __bf16* Q16     = take(nText);
  __bf16* K16     = take(nFeat);

  // 1) cast everything to bf16 (memory-bound pre-passes)
  cvt_f32_to_bf16<<<4096, 256, 0, stream>>>(text, text16, nText / 4);
  cvt_f32_to_bf16<<<2048, 256, 0, stream>>>(feats, feat16, nFeat / 4);
  cvt_f32_to_bf16<<<2048, 256, 0, stream>>>(Wq, Wq16, nW / 4);
  cvt_f32_to_bf16<<<2048, 256, 0, stream>>>(Wk, Wk16, nW / 4);
  // 2) features -> [B][D][P] bf16 so attn@V is an NT WMMA GEMM
  transpose_cvt_bf16<<<dim3(D / 32, P / 32, B), 256, 0, stream>>>(feats, featT16, P, D);
  // 3) projections (scale 1/sqrt(D)=1/64 folded into Q)
  gemm_nt_bias_bf16<<<dim3(D / GBN, (B * T) / GBM), 256, 0, stream>>>(
      text16, Wq16, bq, Q16, D, D, 0.015625f);
  gemm_nt_bias_bf16<<<dim3(D / GBN, (B * P) / GBM), 256, 0, stream>>>(
      feat16, Wk16, bk, K16, D, D, 1.0f);
  // 4) fused scores -> softmax -> attn@V
  cross_attention_core<<<dim3(T / TQ, B), 256, 0, stream>>>(Q16, K16, featT16, out, T, D);
}
